// SimpleDriftingLoss_65111704207366
// MI455X (gfx1250) — matrix operand order repaired and checked by reading, not compiled, not run on hardware
//
#include <hip/hip_runtime.h>
#include <hip/hip_bf16.h>
#include <stdint.h>

// Problem sizes (fixed by the reference): B = N = 8192, D = 128.
#define NB   8192
#define ND   128
#define FEPS 1e-8f

typedef __attribute__((ext_vector_type(16))) _Float16 v16h;
typedef __attribute__((ext_vector_type(8)))  float    v8f;

union Frag16 {            // one WMMA f16 A/B operand (8 VGPRs)
    v16h  v;
    uint4 u[2];
    _Float16 h[16];
};
union FragAcc {           // one WMMA f32 C/D operand (8 VGPRs)
    v8f   v;
    float f[8];
};

// ---- CDNA5 async global->LDS copy (ASYNCcnt-tracked, bypasses VGPRs) ----
__device__ __forceinline__ void async_load_b128_to_lds(const _Float16* g, _Float16* l)
{
    // Generic LDS pointers carry the wave-relative LDS byte offset in addr[31:0].
    unsigned loff = (unsigned)(uintptr_t)l;
    asm volatile("global_load_async_to_lds_b128 %0, %1, off"
                 :: "v"(loff), "v"(g) : "memory");
}
__device__ __forceinline__ void wait_asynccnt0()
{
#if __has_builtin(__builtin_amdgcn_s_wait_asynccnt)
    __builtin_amdgcn_s_wait_asynccnt(0);
#else
    asm volatile("s_wait_asynccnt 0x0" ::: "memory");
#endif
}

__global__ void zero_out_kernel(float* out) { out[0] = 0.0f; }

// One wave (32 lanes) per row: l2-normalize a 128-float row, emit f16.
__global__ __launch_bounds__(128)
void normalize_f16_kernel(const float* __restrict__ x, _Float16* __restrict__ y)
{
    const int lane = threadIdx.x & 31;
    const int wave = threadIdx.x >> 5;
    const int row  = blockIdx.x * 4 + wave;

    float4 v = ((const float4*)(x + (size_t)row * ND))[lane];
    float ss = v.x * v.x + v.y * v.y + v.z * v.z + v.w * v.w;
    #pragma unroll
    for (int m = 16; m >= 1; m >>= 1) ss += __shfl_xor(ss, m, 32);
    const float inv = 1.0f / (sqrtf(ss) + FEPS);

    union { _Float16 h[4]; uint2 u; } o;
    o.h[0] = (_Float16)(v.x * inv);
    o.h[1] = (_Float16)(v.y * inv);
    o.h[2] = (_Float16)(v.z * inv);
    o.h[3] = (_Float16)(v.w * inv);
    ((uint2*)(y + (size_t)row * ND))[lane] = o.u;
}

// Split-K flash pass without max tracking (logits in [-1,1] for unit vectors):
// for key slice s:  Opart[s][q][:] = sum_k exp(q.k) * K[k][:],  Lpart[s][q] = sum_k exp(q.k)
// Block: 128 threads = 4 waves; wave owns 16 query rows; 32 keys per tile;
// key tiles double-buffered in LDS via async global->LDS loads.
__global__ __launch_bounds__(128)
void flash_attn_kernel(const _Float16* __restrict__ Q,
                       const _Float16* __restrict__ K,
                       float* __restrict__ Opart,
                       float* __restrict__ Lpart,
                       int nsplit)
{
    __shared__ __attribute__((aligned(16))) _Float16 lk [2][32 * 128];   // [key][d]
    __shared__ __attribute__((aligned(16))) _Float16 lkT[2][128 * 32];   // [d][key]
    __shared__ __attribute__((aligned(16))) _Float16 lp [4][16 * 32];    // per-wave P [q][key]

    const int lane  = threadIdx.x & 31;
    const int wave  = threadIdx.x >> 5;
    const int n15   = lane & 15;
    const int hi    = lane >> 4;
    const int qbase = blockIdx.x * 64 + wave * 16;
    const int split = blockIdx.y;
    const int tiles_per = (NB / 32) / nsplit;
    const int t0    = split * tiles_per;

    // Preload this wave's Q rows as 4 A-fragments (16x32 f16 A layout:
    // lane m = lane&15; halves 0..7 -> k = hi*8 + h, halves 8..15 -> k = 16 + hi*8 + (h-8)).
    Frag16 qa[4];
    {
        const _Float16* qrow = Q + (size_t)(qbase + n15) * ND;
        #pragma unroll
        for (int f = 0; f < 4; ++f) {
            qa[f].u[0] = *(const uint4*)(qrow + f * 32 + hi * 8);
            qa[f].u[1] = *(const uint4*)(qrow + f * 32 + 16 + hi * 8);
        }
    }

    Frag16 ones;
    #pragma unroll
    for (int i = 0; i < 16; ++i) ones.h[i] = (_Float16)1.0f;

    FragAcc o[8];     // O accumulator, 16 q-rows x 128 dims (8 D-frags)
    FragAcc osum;     // row sums of exp(S), replicated across columns
    #pragma unroll
    for (int j = 0; j < 8; ++j)
        #pragma unroll
        for (int r = 0; r < 8; ++r) o[j].f[r] = 0.0f;
    #pragma unroll
    for (int r = 0; r < 8; ++r) osum.f[r] = 0.0f;

    // Prologue: async-stage tile 0 into buffer 0.
    {
        const _Float16* kt = K + (size_t)t0 * 32 * ND;
        #pragma unroll
        for (int i = 0; i < 4; ++i) {
            const int c   = threadIdx.x + 128 * i;   // 0..511 chunks of 8 halves
            const int key = c >> 4;
            const int dof = (c & 15) * 8;
            async_load_b128_to_lds(kt + key * 128 + dof, &lk[0][key * 128 + dof]);
        }
    }

    #pragma unroll 1
    for (int t = 0; t < tiles_per; ++t) {
        const int buf = t & 1;

        // Wait for this wave's async chunks of tile t, then each thread
        // transposes exactly the chunks it loaded: lk[buf] -> lkT[buf].
        wait_asynccnt0();
        #pragma unroll
        for (int i = 0; i < 4; ++i) {
            const int c   = threadIdx.x + 128 * i;
            const int key = c >> 4;
            const int dof = (c & 15) * 8;
            union { uint4 u; _Float16 h[8]; } ld;
            ld.u = *(const uint4*)(&lk[buf][key * 128 + dof]);
            #pragma unroll
            for (int j = 0; j < 8; ++j) lkT[buf][(dof + j) * 32 + key] = ld.h[j];
        }
        // Overlap: kick off async stage of tile t+1 into the other buffer.
        if (t + 1 < tiles_per) {
            const _Float16* kt = K + (size_t)(t0 + t + 1) * 32 * ND;
            #pragma unroll
            for (int i = 0; i < 4; ++i) {
                const int c   = threadIdx.x + 128 * i;
                const int key = c >> 4;
                const int dof = (c & 15) * 8;
                async_load_b128_to_lds(kt + key * 128 + dof, &lk[buf ^ 1][key * 128 + dof]);
            }
        }
        __syncthreads();   // tile t fully staged+transposed by all waves

        // ---- S = Q . K^T for two 16-key subtiles; P = exp(S) -> wave LDS ----
        #pragma unroll
        for (int s = 0; s < 2; ++s) {
            // Batch-preload all 4 B-fragments of this subtile so the backend
            // can overlap LDS latency with the WMMA accumulate chain.
            Frag16 bk[4];
            #pragma unroll
            for (int f = 0; f < 4; ++f) {
                const _Float16* base = &lk[buf][(s * 16 + n15) * 128 + f * 32 + hi * 8];
                bk[f].u[0] = *(const uint4*)(base);
                bk[f].u[1] = *(const uint4*)(base + 16);
            }
            FragAcc c;
            #pragma unroll
            for (int r = 0; r < 8; ++r) c.f[r] = 0.0f;
            #pragma unroll
            for (int f = 0; f < 4; ++f)
                c.v = __builtin_amdgcn_wmma_f32_16x16x32_f16(
                        false, qa[f].v, false, bk[f].v, (short)0, c.v, false, false);
            // D layout: lane col n = key = s*16+n15; VGPR r -> q-row = r + 8*hi
            #pragma unroll
            for (int r = 0; r < 8; ++r) {
                const float e = __expf(c.f[r]);
                lp[wave][(r + 8 * hi) * 32 + s * 16 + n15] = (_Float16)e;
            }
        }

        // ---- read P back as an A-fragment (16 q x 32 key) ----
        Frag16 pa;
        {
            const _Float16* prow = &lp[wave][n15 * 32];
            __builtin_memcpy(&pa.h[0], prow + hi * 8, 16);
            __builtin_memcpy(&pa.h[8], prow + 16 + hi * 8, 16);
        }

        // ---- O += P . V  (V = key tile), plus row-sum via all-ones B ----
        #pragma unroll
        for (int g = 0; g < 2; ++g) {
            Frag16 bv[4];
            #pragma unroll
            for (int j = 0; j < 4; ++j) {
                const _Float16* base = &lkT[buf][((g * 4 + j) * 16 + n15) * 32 + hi * 8];
                bv[j].u[0] = *(const uint4*)(base);
                bv[j].u[1] = *(const uint4*)(base + 16);
            }
            #pragma unroll
            for (int j = 0; j < 4; ++j)
                o[g * 4 + j].v = __builtin_amdgcn_wmma_f32_16x16x32_f16(
                        false, pa.v, false, bv[j].v, (short)0, o[g * 4 + j].v, false, false);
        }
        osum.v = __builtin_amdgcn_wmma_f32_16x16x32_f16(
                    false, pa.v, false, ones.v, (short)0, osum.v, false, false);

        __syncthreads();   // everyone done with buffers before they are re-staged
    }

    // ---- store partials (unnormalized) ----
    #pragma unroll
    for (int r = 0; r < 8; ++r) {
        const int q = qbase + r + 8 * hi;
        float* orow = Opart + ((size_t)split * NB + q) * ND;
        #pragma unroll
        for (int j = 0; j < 8; ++j)
            orow[j * 16 + n15] = o[j].f[r];
        if (n15 == 0)
            Lpart[(size_t)split * NB + q] = osum.f[r];
    }
}

// Merge splits, form V = pos - neg per row, accumulate mean-square loss.
__global__ __launch_bounds__(128)
void combine_loss_kernel(const float* __restrict__ Op, const float* __restrict__ Lp,
                         const float* __restrict__ On, const float* __restrict__ Ln,
                         float* __restrict__ out, int nsplit)
{
    const int lane = threadIdx.x & 31;
    const int wave = threadIdx.x >> 5;
    const int row  = blockIdx.x * 4 + wave;

    float px = 0.f, py = 0.f, pz = 0.f, pw = 0.f;
    float nx = 0.f, ny = 0.f, nz = 0.f, nw = 0.f;
    float lpos = 0.f, lneg = 0.f;
    for (int s = 0; s < nsplit; ++s) {
        const float4 a = ((const float4*)(Op + ((size_t)s * NB + row) * ND))[lane];
        const float4 b = ((const float4*)(On + ((size_t)s * NB + row) * ND))[lane];
        px += a.x; py += a.y; pz += a.z; pw += a.w;
        nx += b.x; ny += b.y; nz += b.z; nw += b.w;
        lpos += Lp[(size_t)s * NB + row];
        lneg += Ln[(size_t)s * NB + row];
    }
    const float ip = 1.0f / lpos, in = 1.0f / lneg;
    const float dx = px * ip - nx * in;
    const float dy = py * ip - ny * in;
    const float dz = pz * ip - nz * in;
    const float dw = pw * ip - nw * in;
    float ss = dx * dx + dy * dy + dz * dz + dw * dw;
    #pragma unroll
    for (int m = 16; m >= 1; m >>= 1) ss += __shfl_xor(ss, m, 32);

    if (lane == 0) {
        const float nm = sqrtf(ss) + FEPS;
        const float contrib = ss / (nm * nm);
        atomicAdd(out, contrib * (1.0f / ((float)NB * (float)ND)));
    }
}

extern "C" void kernel_launch(void* const* d_in, const int* in_sizes, int n_in,
                              void* d_out, int out_size, void* d_ws, size_t ws_size,
                              hipStream_t stream)
{
    const float* x_gen  = (const float*)d_in[0];   // (8192, 128) f32
    const float* x_data = (const float*)d_in[1];   // (8192, 128) f32
    float* out = (float*)d_out;                    // scalar f32

    // Choose split-K factor deterministically from ws_size.
    const size_t base = (size_t)4 * 1024 * 1024;   // Yn + Xn (f16)
    int nsplit = 1;
    for (int s = 8; s >= 2; s >>= 1) {
        const size_t need = base + 2 * ((size_t)s * NB * ND * 4 + (size_t)s * NB * 4);
        if (need <= ws_size) { nsplit = s; break; }
    }

    char* ws = (char*)d_ws;
    _Float16* Yn = (_Float16*)(ws);
    _Float16* Xn = (_Float16*)(ws + (size_t)2 * 1024 * 1024);
    const size_t sizeO = (size_t)nsplit * NB * ND;   // floats
    const size_t sizeL = (size_t)nsplit * NB;        // floats
    float* Opos = (float*)(ws + base);
    float* Oneg = Opos + sizeO;
    float* Lpos = Oneg + sizeO;
    float* Lneg = Lpos + sizeL;

    zero_out_kernel<<<1, 1, 0, stream>>>(out);
    normalize_f16_kernel<<<NB / 4, 128, 0, stream>>>(x_gen,  Yn);
    normalize_f16_kernel<<<NB / 4, 128, 0, stream>>>(x_data, Xn);

    dim3 fgrid(NB / 64, nsplit);
    flash_attn_kernel<<<fgrid, 128, 0, stream>>>(Yn, Xn, Opos, Lpos, nsplit); // pos pass
    flash_attn_kernel<<<fgrid, 128, 0, stream>>>(Yn, Yn, Oneg, Lneg, nsplit); // neg pass

    combine_loss_kernel<<<NB / 4, 128, 0, stream>>>(Opos, Lpos, Oneg, Lneg, out, nsplit);
}
